// Attention_22634477650597
// MI455X (gfx1250) — matrix operand order repaired and checked by reading, not compiled, not run on hardware
//
#include <hip/hip_runtime.h>
#include <hip/hip_bf16.h>
#include <math.h>

typedef __bf16 bf16_t;
typedef bf16_t v16bf __attribute__((ext_vector_type(16)));
typedef bf16_t v8bf  __attribute__((ext_vector_type(8)));
typedef float  v8f   __attribute__((ext_vector_type(8)));

#define B_   16
#define S_   4096
#define D2_  1024   // 2H
#define H_   512
#define MTILE 64    // s-rows per workgroup (4 M-tiles of 16 per wave)
#define LDSTR 1032  // 1024 + 8 bf16 pad -> conflict-free ds_load_b128 across rows

// ---------- kernel 0: W_bot (rows D2_..2*D2_-1 of W_attn [2048,512]) -> bf16, transposed [H][D2] ----------
__global__ void k_convert_w(const float* __restrict__ W_attn, bf16_t* __restrict__ WbT) {
    int idx = blockIdx.x * blockDim.x + threadIdx.x;     // idx = n*D2 + k
    if (idx >= H_ * D2_) return;
    int k = idx & (D2_ - 1);
    int n = idx >> 10;
    WbT[idx] = (bf16_t)W_attn[(size_t)(D2_ + k) * H_ + n];
}

// ---------- kernel 1: hproj[b][h] = hidden[b] @ W_top[:,h] + b_attn[h] ----------
__global__ void k_hproj(const float* __restrict__ hidden, const float* __restrict__ W_attn,
                        const float* __restrict__ b_attn, float* __restrict__ hproj) {
    int idx = blockIdx.x * blockDim.x + threadIdx.x;     // idx = b*H + h
    if (idx >= B_ * H_) return;
    int h = idx & (H_ - 1);
    int b = idx >> 9;
    float acc = b_attn[h];
    const float* hid = hidden + (size_t)b * D2_;
    for (int k = 0; k < D2_; ++k)
        acc = fmaf(hid[k], W_attn[(size_t)k * H_ + h], acc);
    hproj[idx] = acc;
}

// ---------- kernel 2: scores[b][s] = w_v . tanh(enc[b,s] @ W_bot + hproj[b]) ----------
// Grid: (S/64, B). Block: 256 (8 waves). Wave = 4 M-tiles x 4 N-tiles -> 16 WMMA / K-step.
__global__ void __launch_bounds__(256)
k_scores(const float* __restrict__ enc, const bf16_t* __restrict__ WbT,
         const float* __restrict__ hproj, const float* __restrict__ w_v,
         float* __restrict__ scores) {
    __shared__ bf16_t Atile[MTILE * LDSTR];   // ~132 KB of the 320 KB WGP LDS
    __shared__ float  sacc[MTILE];

    const int tid  = threadIdx.x;
    const int lane = tid & 31;
    const int wave = tid >> 5;
    const int b    = blockIdx.y;
    const int s0   = blockIdx.x * MTILE;

    if (tid < MTILE) sacc[tid] = 0.0f;

    // Stage the 64x1024 f32 encoder tile (contiguous in memory) into LDS as bf16.
    const float* src = enc + ((size_t)b * S_ + s0) * D2_;
    for (int i = tid * 4; i < MTILE * D2_; i += 256 * 4) {
        const float4 f = *(const float4*)(src + i);
        int row = i >> 10, col = i & (D2_ - 1);
        bf16_t* d = &Atile[row * LDSTR + col];
        d[0] = (bf16_t)f.x; d[1] = (bf16_t)f.y; d[2] = (bf16_t)f.z; d[3] = (bf16_t)f.w;
    }
    __syncthreads();

    const int hi = lane >> 4;      // half-wave select
    const int nl = lane & 15;      // row (A) / column (B) within a tile

    float hb[4], wv[4];
#pragma unroll
    for (int t = 0; t < 4; ++t) {
        int h = (wave * 4 + t) * 16 + nl;
        hb[t] = hproj[b * H_ + h];
        wv[t] = w_v[h];
    }

    v8f acc[4][4];                 // [m-tile][n-tile], 128 VGPRs
#pragma unroll
    for (int m = 0; m < 4; ++m)
#pragma unroll
        for (int t = 0; t < 4; ++t)
#pragma unroll
            for (int i = 0; i < 8; ++i) acc[m][t][i] = 0.0f;

    const int kA = hi * 8;    // A lane-half K offset: {0..7,16..23} / {8..15,24..31}
    const int kB = hi * 8;    // B lane-half K offset: {0..7,16..23} / {8..15,24..31}

    for (int kc = 0; kc < D2_; kc += 32) {
        // One set of B fragments per wave, reused across all 4 M-tiles.
        v16bf bfrag[4];
#pragma unroll
        for (int t = 0; t < 4; ++t) {
            int n = (wave * 4 + t) * 16 + nl;
            const bf16_t* bp = &WbT[(size_t)n * D2_ + kc + kB];
            v8bf blo = *(const v8bf*)bp;
            v8bf bhi = *(const v8bf*)(bp + 16);
            bfrag[t] = __builtin_shufflevector(blo, bhi, 0,1,2,3,4,5,6,7,8,9,10,11,12,13,14,15);
            if (kc + 32 < D2_) __builtin_prefetch(bp + 32, 0, 3);
        }
#pragma unroll
        for (int m = 0; m < 4; ++m) {
            const bf16_t* ap = &Atile[(m * 16 + nl) * LDSTR + kc + kA];
            v8bf alo = *(const v8bf*)ap;
            v8bf ahi = *(const v8bf*)(ap + 16);
            v16bf afrag = __builtin_shufflevector(alo, ahi, 0,1,2,3,4,5,6,7,8,9,10,11,12,13,14,15);
#pragma unroll
            for (int t = 0; t < 4; ++t)
                acc[m][t] = __builtin_amdgcn_wmma_f32_16x16x32_bf16(
                    false, afrag, false, bfrag[t], (short)0, acc[m][t], false, false);
        }
    }

    // Epilogue per M-tile: acc[m][t][i] = energy[row][h], row = m*16 + i + 8*hi.
#pragma unroll
    for (int m = 0; m < 4; ++m) {
        float part[8];
#pragma unroll
        for (int i = 0; i < 8; ++i) part[i] = 0.0f;
#pragma unroll
        for (int t = 0; t < 4; ++t)
#pragma unroll
            for (int i = 0; i < 8; ++i)
                part[i] += tanhf(acc[m][t][i] + hb[t]) * wv[t];

        // Reduce over the 16 h-lanes of each half-wave.
#pragma unroll
        for (int off = 1; off < 16; off <<= 1)
#pragma unroll
            for (int i = 0; i < 8; ++i)
                part[i] += __shfl_xor(part[i], off, 32);

        if (nl == 0) {
#pragma unroll
            for (int i = 0; i < 8; ++i)
                atomicAdd(&sacc[m * 16 + i + 8 * hi], part[i]);   // ds_add_f32 across waves
        }
    }
    __syncthreads();
    if (tid < MTILE) scores[(size_t)b * S_ + s0 + tid] = sacc[tid];
}

// ---------- kernel 3: softmax over S per batch (in place) ----------
__global__ void k_softmax(float* __restrict__ scores) {
    __shared__ float red[256];
    const int b = blockIdx.x, tid = threadIdx.x;
    float* row = scores + (size_t)b * S_;

    float m = -INFINITY;
    for (int s = tid; s < S_; s += 256) m = fmaxf(m, row[s]);
    red[tid] = m; __syncthreads();
    for (int off = 128; off > 0; off >>= 1) {
        if (tid < off) red[tid] = fmaxf(red[tid], red[tid + off]);
        __syncthreads();
    }
    m = red[0]; __syncthreads();

    float sum = 0.0f;
    for (int s = tid; s < S_; s += 256) {
        float e = __expf(row[s] - m);
        row[s] = e;
        sum += e;
    }
    red[tid] = sum; __syncthreads();
    for (int off = 128; off > 0; off >>= 1) {
        if (tid < off) red[tid] += red[tid + off];
        __syncthreads();
    }
    float inv = 1.0f / red[0];
    for (int s = tid; s < S_; s += 256) row[s] *= inv;
}

// ---------- kernel 4a: zero the output (atomics accumulate into it) ----------
__global__ void k_zero(float* __restrict__ out, int n) {
    int i = blockIdx.x * blockDim.x + threadIdx.x;
    if (i < n) out[i] = 0.0f;
}

// ---------- kernel 4b: context[b][d] = sum_s attn[b][s] * enc[b][s][d] ----------
// Grid: (32 s-chunks of 128, B). Block 256: each thread owns 4 contiguous d-columns.
__global__ void k_context(const float* __restrict__ enc, const float* __restrict__ attn,
                          float* __restrict__ out) {
    const int b = blockIdx.y;
    const int c = blockIdx.x;             // s-chunk of 128
    const int d = threadIdx.x * 4;
    float4 acc = make_float4(0.f, 0.f, 0.f, 0.f);
    const float* ebase = enc + ((size_t)b * S_ + (size_t)c * 128) * D2_;
    const float* arow  = attn + (size_t)b * S_ + (size_t)c * 128;
    for (int s = 0; s < 128; ++s) {
        float a = arow[s];
        float4 e = *(const float4*)(ebase + (size_t)s * D2_ + d);
        acc.x = fmaf(a, e.x, acc.x);
        acc.y = fmaf(a, e.y, acc.y);
        acc.z = fmaf(a, e.z, acc.z);
        acc.w = fmaf(a, e.w, acc.w);
    }
    float* o = out + (size_t)b * D2_ + d;
    atomicAdd(o + 0, acc.x);
    atomicAdd(o + 1, acc.y);
    atomicAdd(o + 2, acc.z);
    atomicAdd(o + 3, acc.w);
}

extern "C" void kernel_launch(void* const* d_in, const int* in_sizes, int n_in,
                              void* d_out, int out_size, void* d_ws, size_t ws_size,
                              hipStream_t stream) {
    const float* hidden = (const float*)d_in[0];   // [B, 2H]
    const float* enc    = (const float*)d_in[1];   // [B, S, 2H]
    const float* W_attn = (const float*)d_in[2];   // [4H, H]
    const float* b_attn = (const float*)d_in[3];   // [H]
    const float* w_v    = (const float*)d_in[4];   // [H]
    float* out = (float*)d_out;                    // [B, 2H]

    char* ws = (char*)d_ws;
    bf16_t* WbT    = (bf16_t*)ws;                             // 512*1024*2 = 1 MB
    float*  hproj  = (float*)(ws + (1u << 20));               // 16*512*4   = 32 KB
    float*  scores = (float*)(ws + (1u << 20) + (32u << 10)); // 16*4096*4  = 256 KB

    k_convert_w<<<(H_ * D2_ + 255) / 256, 256, 0, stream>>>(W_attn, WbT);
    k_hproj<<<(B_ * H_ + 255) / 256, 256, 0, stream>>>(hidden, W_attn, b_attn, hproj);
    k_scores<<<dim3(S_ / MTILE, B_), 256, 0, stream>>>(enc, WbT, hproj, w_v, scores);
    k_softmax<<<B_, 256, 0, stream>>>(scores);
    k_zero<<<(B_ * D2_ + 255) / 256, 256, 0, stream>>>(out, B_ * D2_);
    k_context<<<dim3(S_ / 128, B_), 256, 0, stream>>>(enc, scores, out);
}
